// MambaBackbone_30159260352533
// MI455X (gfx1250) — matrix operand order repaired and checked by reading, not compiled, not run on hardware
//
#include <hip/hip_runtime.h>
#include <hip/hip_bf16.h>
#include <math.h>

// ---------------- model constants ----------------
#define D_MODEL  1024
#define D_INNER  2048
#define D_STATE  16
#define DT_RANK  64
#define N_LAYERS 24
#define SEQ      577           // 1 cls + 24*24 patches
#define NBATCH   2
#define MROWS    (NBATCH*SEQ)  // 1154
#define NPATCH   (NBATCH*576)  // 1152
#define KPATCH   768           // 3*16*16

typedef __attribute__((ext_vector_type(16))) __bf16 bf16x16;
typedef __attribute__((ext_vector_type(8)))  __bf16 bf16x8;
typedef __attribute__((ext_vector_type(8)))  float  f32x8;

__device__ __forceinline__ float silu_f(float x)    { return x / (1.f + __expf(-x)); }
__device__ __forceinline__ float softplus_f(float x){ return (x > 20.f) ? x : log1pf(__expf(x)); }

// ---------------- WMMA GEMM: Out[M,N] = A[M,K(lda)] * W[N,K(ldw)]^T (+epilogue) ----
// Block tile 128x128, K-step 64. 8 waves as 4(M) x 2(N); each wave owns 32x64
// (2 M-frags x 4 N-frags = 8 f32x8 accumulators, 16 v_wmma per K-step).
// mode 0: plain   1: +bias[n]   2: softplus(acc+bias[n])   3: residual add into Out
#define BM 128
#define BN 128
#define BK 64
#define LDSS 72   // bf16 elems per LDS row (144B: 16B-aligned, bank-skewed)

__global__ __launch_bounds__(256) void wmma_gemm_kernel(
    const float* __restrict__ A, int lda,
    const float* __restrict__ W, int ldw,
    float* __restrict__ Out, int ldo,
    const float* __restrict__ bias,
    int M, int N, int K, int mode)
{
  __shared__ __bf16 As[BM * LDSS];
  __shared__ __bf16 Ws[BN * LDSS];

  const int tid  = threadIdx.x;
  const int lane = tid & 31;
  const int wave = tid >> 5;
  const int wm   = wave & 3;   // 4 waves tile M (4*32 = 128)
  const int wn   = wave >> 2;  // 2 waves tile N (2*64 = 128)
  const int m0   = blockIdx.y * BM;
  const int n0   = blockIdx.x * BN;

  f32x8 acc[2][4];
  #pragma unroll
  for (int mi = 0; mi < 2; ++mi)
    #pragma unroll
    for (int ni = 0; ni < 4; ++ni) acc[mi][ni] = (f32x8){};

  // staging: 256 threads cover 128 rows x 64 cols; 2 threads/row, 32 cols each
  const int ldRow = tid >> 1;        // 0..127
  const int ldCol = (tid & 1) * 32;  // 0 or 32
  // clamp instead of branch: OOB rows compute garbage that the epilogue discards
  int gr = m0 + ldRow; if (gr >= M) gr = M - 1;
  int gn = n0 + ldRow; if (gn >= N) gn = N - 1;
  const float* aSrcBase = A + (size_t)gr * lda + ldCol;
  const float* wSrcBase = W + (size_t)gn * ldw + ldCol;
  __bf16* aDst = &As[ldRow * LDSS + ldCol];
  __bf16* wDst = &Ws[ldRow * LDSS + ldCol];

  const int nk = K / BK;
  for (int kt = 0; kt < nk; ++kt) {
    const int k0 = kt * BK;
    {
      const float* src = aSrcBase + k0;
      #pragma unroll
      for (int j = 0; j < 32; ++j) aDst[j] = (__bf16)src[j];
    }
    {
      const float* src = wSrcBase + k0;
      #pragma unroll
      for (int j = 0; j < 32; ++j) wDst[j] = (__bf16)src[j];
    }
    if (kt + 1 < nk) {  // lowers to global_prefetch_b8
      __builtin_prefetch(aSrcBase + k0 + BK, 0, 0);
      __builtin_prefetch(wSrcBase + k0 + BK, 0, 0);
    }
    __syncthreads();

    #pragma unroll
    for (int ks = 0; ks < 2; ++ks) {
      const int kk = ks * 32;
      // A frags: lanes 0-15 -> K kk+0..7 & kk+16..23 ; lanes 16-31 -> +8
      const int kb = (lane >> 4) * 8;
      bf16x16 af[2];
      #pragma unroll
      for (int mi = 0; mi < 2; ++mi) {
        const int arow = wm * 32 + mi * 16 + (lane & 15);
        bf16x8 lo = *(const bf16x8*)&As[arow * LDSS + kk + kb];
        bf16x8 hi = *(const bf16x8*)&As[arow * LDSS + kk + 16 + kb];
        af[mi] = __builtin_shufflevector(lo, hi,
            0,1,2,3,4,5,6,7,8,9,10,11,12,13,14,15);
      }
      // B frags: lane holds col N=lane&15; K = kk + (lane>=16?8:0) + 0..7 & +16..23
      const int kb2 = kk + (lane >> 4) * 8;
      #pragma unroll
      for (int ni = 0; ni < 4; ++ni) {
        const int brow = wn * 64 + ni * 16 + (lane & 15);
        bf16x8 lo = *(const bf16x8*)&Ws[brow * LDSS + kb2];
        bf16x8 hi = *(const bf16x8*)&Ws[brow * LDSS + kb2 + 16];
        bf16x16 bf = __builtin_shufflevector(lo, hi,
            0,1,2,3,4,5,6,7,8,9,10,11,12,13,14,15);
        acc[0][ni] = __builtin_amdgcn_wmma_f32_16x16x32_bf16(
            false, af[0], false, bf, (short)0, acc[0][ni], false, false);
        acc[1][ni] = __builtin_amdgcn_wmma_f32_16x16x32_bf16(
            false, af[1], false, bf, (short)0, acc[1][ni], false, false);
      }
    }
    __syncthreads();
  }

  // epilogue: C/D layout — VGPR r holds M = r + 8*(lane>=16), N = lane&15
  const int eroff = (lane >> 4) * 8;
  const int ecol  = lane & 15;
  #pragma unroll
  for (int mi = 0; mi < 2; ++mi) {
    #pragma unroll
    for (int r = 0; r < 8; ++r) {
      const int row = m0 + wm * 32 + mi * 16 + eroff + r;
      if (row >= M) continue;
      #pragma unroll
      for (int ni = 0; ni < 4; ++ni) {
        const int col = n0 + wn * 64 + ni * 16 + ecol;
        if (col >= N) continue;
        float v = acc[mi][ni][r];
        size_t o = (size_t)row * ldo + col;
        if      (mode == 1) Out[o] = v + bias[col];
        else if (mode == 2) Out[o] = softplus_f(v + bias[col]);
        else if (mode == 3) Out[o] = Out[o] + v;
        else                Out[o] = v;
      }
    }
  }
}

// ---------------- im2col for 16x16 stride-16 patch embed ----------------
__global__ void im2col_kernel(const float* __restrict__ x, float* __restrict__ col)
{
  int idx = blockIdx.x * blockDim.x + threadIdx.x;   // NPATCH*KPATCH threads
  int row = idx / KPATCH;
  int c_  = idx % KPATCH;
  int b   = row / 576;
  int hw  = row % 576;
  int h   = hw / 24, w = hw % 24;
  int c   = c_ / 256;
  int r2  = c_ % 256;
  int ph  = r2 / 16, pw = r2 % 16;
  col[idx] = x[(((size_t)b * 3 + c) * 384 + (h * 16 + ph)) * 384 + (w * 16 + pw)];
}

// ---------------- tokens = [cls | patches] + pos_embed (resize identity 24->24) ----
__global__ void assemble_tokens_kernel(const float* __restrict__ feat,
                                       const float* __restrict__ cls,
                                       const float* __restrict__ pos,
                                       float* __restrict__ tok)
{
  int idx = blockIdx.x * blockDim.x + threadIdx.x;   // NBATCH*SEQ*D_MODEL
  int d = idx % D_MODEL;
  int t = (idx / D_MODEL) % SEQ;
  int b = idx / (D_MODEL * SEQ);
  float v;
  if (t == 0) v = cls[d] + pos[d];
  else        v = feat[((size_t)b * 576 + (t - 1)) * D_MODEL + d] + pos[(size_t)t * D_MODEL + d];
  tok[idx] = v;
}

// ---------------- LayerNorm (one block per token row) ----------------
__global__ __launch_bounds__(256) void layernorm_kernel(
    const float* __restrict__ X, const float* __restrict__ w,
    const float* __restrict__ bb, float* __restrict__ Y)
{
  __shared__ float red[256];
  const int m = blockIdx.x;
  const int t = threadIdx.x;
  const float* x = X + (size_t)m * D_MODEL;
  float s = 0.f;
  for (int i = t; i < D_MODEL; i += 256) s += x[i];
  red[t] = s; __syncthreads();
  for (int o = 128; o > 0; o >>= 1) { if (t < o) red[t] += red[t + o]; __syncthreads(); }
  const float mean = red[0] * (1.f / D_MODEL);
  __syncthreads();
  float v = 0.f;
  for (int i = t; i < D_MODEL; i += 256) { float d = x[i] - mean; v += d * d; }
  red[t] = v; __syncthreads();
  for (int o = 128; o > 0; o >>= 1) { if (t < o) red[t] += red[t + o]; __syncthreads(); }
  const float rstd = rsqrtf(red[0] * (1.f / D_MODEL) + 1e-5f);
  float* y = Y + (size_t)m * D_MODEL;
  for (int i = t; i < D_MODEL; i += 256) y[i] = (x[i] - mean) * rstd * w[i] + bb[i];
}

// ---------------- depthwise causal conv (k=4, pad 3 left) + SiLU ----------------
__global__ void conv_silu_kernel(const float* __restrict__ xz,
                                 const float* __restrict__ cw,
                                 const float* __restrict__ cb,
                                 float* __restrict__ U)
{
  int idx = blockIdx.x * blockDim.x + threadIdx.x;   // MROWS*D_INNER
  int d = idx & (D_INNER - 1);
  int m = idx >> 11;
  int b = m / SEQ, l = m % SEQ;
  float acc = cb[d];
  #pragma unroll
  for (int k = 0; k < 4; ++k) {
    int lp = l + k - 3;
    if (lp >= 0)
      acc += cw[d * 4 + k] * xz[(size_t)(b * SEQ + lp) * (2 * D_INNER) + d];
  }
  U[(size_t)m * D_INNER + d] = silu_f(acc);
}

// ---------------- selective scan: thread per (batch, channel), 16 states in regs ----
// Per-step B/C broadcast (32 contiguous f32 = 8 x b128) is staged global->LDS with
// the CDNA5 async-LDS engine (global_load_async_to_lds_b128 / s_wait_asynccnt),
// double-buffered one step ahead of the recurrence.
__global__ __launch_bounds__(256) void scan_kernel(
    const float* __restrict__ dt, const float* __restrict__ u,
    const float* __restrict__ dbl, const float* __restrict__ alog,
    float* __restrict__ ys)
{
  __shared__ float sBC[2][32];   // [buf][ B(0..15) | C(16..31) ]
  const int d = blockIdx.x * 256 + threadIdx.x;
  const int b = blockIdx.y;
  float Areg[D_STATE], h[D_STATE];
  #pragma unroll
  for (int s = 0; s < D_STATE; ++s) {
    Areg[s] = -__expf(alog[(size_t)d * D_STATE + s]);
    h[s] = 0.f;
  }
  // prologue: async-copy B/C of step 0 into buffer 0 (lanes 0..7, 16B each)
  if (threadIdx.x < 8) {
    const float* gsrc = dbl + ((size_t)b * SEQ) * 96 + DT_RANK;
    unsigned ldsAddr = (unsigned)(unsigned long long)(const void*)&sBC[0][threadIdx.x * 4];
    unsigned voff = threadIdx.x * 16u;
    asm volatile("global_load_async_to_lds_b128 %0, %1, %2"
                 :: "v"(ldsAddr), "v"(voff), "s"((unsigned long long)gsrc)
                 : "memory");
  }
  for (int l = 0; l < SEQ; ++l) {
    asm volatile("s_wait_asynccnt 0x0" ::: "memory");
    __syncthreads();
    const int buf = l & 1;
    const size_t m = (size_t)b * SEQ + l;
    const float dtv = dt[m * D_INNER + d];
    const float du  = dtv * u[m * D_INNER + d];
    float y = 0.f;
    #pragma unroll
    for (int s = 0; s < D_STATE; ++s) {
      h[s] = __expf(dtv * Areg[s]) * h[s] + du * sBC[buf][s];
      y += h[s] * sBC[buf][D_STATE + s];
    }
    ys[m * D_INNER + d] = y;
    __syncthreads();   // everyone done reading buf before it is re-targeted
    if (l + 1 < SEQ && threadIdx.x < 8) {
      const float* gsrc = dbl + ((size_t)b * SEQ + (l + 1)) * 96 + DT_RANK;
      unsigned ldsAddr = (unsigned)(unsigned long long)(const void*)&sBC[(l + 1) & 1][threadIdx.x * 4];
      unsigned voff = threadIdx.x * 16u;
      asm volatile("global_load_async_to_lds_b128 %0, %1, %2"
                   :: "v"(ldsAddr), "v"(voff), "s"((unsigned long long)gsrc)
                   : "memory");
    }
  }
}

// ---------------- gate: y = (ys + D_skip*u) * silu(z) (in place) ----------------
__global__ void gate_kernel(float* __restrict__ y, const float* __restrict__ u,
                            const float* __restrict__ xz, const float* __restrict__ dsk)
{
  int idx = blockIdx.x * blockDim.x + threadIdx.x;   // MROWS*D_INNER
  int d = idx & (D_INNER - 1);
  int m = idx >> 11;
  float z = xz[(size_t)m * (2 * D_INNER) + D_INNER + d];
  y[idx] = (y[idx] + dsk[d] * u[idx]) * silu_f(z);
}

// ---------------- host-side orchestration ----------------
extern "C" void kernel_launch(void* const* d_in, const int* in_sizes, int n_in,
                              void* d_out, int out_size, void* d_ws, size_t ws_size,
                              hipStream_t stream)
{
  (void)in_sizes; (void)n_in; (void)out_size; (void)ws_size;
  const float* x         = (const float*)d_in[0];
  const float* patch_w   = (const float*)d_in[1];
  const float* patch_b   = (const float*)d_in[2];
  const float* cls_tok   = (const float*)d_in[3];
  const float* pos_emb   = (const float*)d_in[4];
  const float* ln_w      = (const float*)d_in[5];
  const float* ln_b      = (const float*)d_in[6];
  const float* in_proj_w = (const float*)d_in[7];
  const float* conv_w    = (const float*)d_in[8];
  const float* conv_b    = (const float*)d_in[9];
  const float* x_proj_w  = (const float*)d_in[10];
  const float* dt_proj_w = (const float*)d_in[11];
  const float* dt_proj_b = (const float*)d_in[12];
  const float* A_log     = (const float*)d_in[13];
  const float* D_skip    = (const float*)d_in[14];
  const float* out_proj_w= (const float*)d_in[15];

  float* tokens = (float*)d_out;                    // [2,577,1024] final output
  float* ws     = (float*)d_ws;
  float* im2col = ws;                                           // 1152*768
  float* feat   = im2col + (size_t)NPATCH * KPATCH;             // 1152*1024
  float* hnorm  = feat   + (size_t)NPATCH * D_MODEL;            // 1154*1024
  float* xz     = hnorm  + (size_t)MROWS * D_MODEL;             // 1154*4096
  float* ubuf   = xz     + (size_t)MROWS * 2 * D_INNER;         // 1154*2048
  float* dbl    = ubuf   + (size_t)MROWS * D_INNER;             // 1154*96
  float* dtbuf  = dbl    + (size_t)MROWS * 96;                  // 1154*2048
  float* ybuf   = dtbuf  + (size_t)MROWS * D_INNER;             // 1154*2048

  // ---- patch embed as GEMM ----
  im2col_kernel<<<(NPATCH * KPATCH) / 256, 256, 0, stream>>>(x, im2col);
  {
    dim3 g((D_MODEL + BN - 1) / BN, (NPATCH + BM - 1) / BM);
    wmma_gemm_kernel<<<g, 256, 0, stream>>>(im2col, KPATCH, patch_w, KPATCH,
                                            feat, D_MODEL, patch_b,
                                            NPATCH, D_MODEL, KPATCH, /*mode*/1);
  }
  assemble_tokens_kernel<<<(NBATCH * SEQ * D_MODEL) / 256, 256, 0, stream>>>(
      feat, cls_tok, pos_emb, tokens);

  // ---- 24 Mamba layers ----
  for (int layer = 0; layer < N_LAYERS; ++layer) {
    const float* lw  = ln_w      + (size_t)layer * D_MODEL;
    const float* lb  = ln_b      + (size_t)layer * D_MODEL;
    const float* ipw = in_proj_w + (size_t)layer * 2 * D_INNER * D_MODEL;
    const float* cw  = conv_w    + (size_t)layer * D_INNER * 4;
    const float* cb  = conv_b    + (size_t)layer * D_INNER;
    const float* xpw = x_proj_w  + (size_t)layer * 96 * D_INNER;
    const float* dpw = dt_proj_w + (size_t)layer * D_INNER * DT_RANK;
    const float* dpb = dt_proj_b + (size_t)layer * D_INNER;
    const float* alg = A_log     + (size_t)layer * D_INNER * D_STATE;
    const float* dsk = D_skip    + (size_t)layer * D_INNER;
    const float* opw = out_proj_w+ (size_t)layer * D_MODEL * D_INNER;

    layernorm_kernel<<<MROWS, 256, 0, stream>>>(tokens, lw, lb, hnorm);

    { // in_proj: [1154,1024] x [4096,1024]^T -> xz [1154,4096]
      dim3 g((2 * D_INNER + BN - 1) / BN, (MROWS + BM - 1) / BM);
      wmma_gemm_kernel<<<g, 256, 0, stream>>>(hnorm, D_MODEL, ipw, D_MODEL,
                                              xz, 2 * D_INNER, nullptr,
                                              MROWS, 2 * D_INNER, D_MODEL, 0);
    }
    conv_silu_kernel<<<(MROWS * D_INNER) / 256, 256, 0, stream>>>(xz, cw, cb, ubuf);

    { // x_proj: [1154,2048] x [96,2048]^T -> dbl [1154,96]
      dim3 g((96 + BN - 1) / BN, (MROWS + BM - 1) / BM);
      wmma_gemm_kernel<<<g, 256, 0, stream>>>(ubuf, D_INNER, xpw, D_INNER,
                                              dbl, 96, nullptr,
                                              MROWS, 96, D_INNER, 0);
    }
    { // dt_proj: [1154,64] x [2048,64]^T + bias, softplus -> dt [1154,2048]
      dim3 g((D_INNER + BN - 1) / BN, (MROWS + BM - 1) / BM);
      wmma_gemm_kernel<<<g, 256, 0, stream>>>(dbl, 96, dpw, DT_RANK,
                                              dtbuf, D_INNER, dpb,
                                              MROWS, D_INNER, DT_RANK, 2);
    }
    { // selective scan
      dim3 g(D_INNER / 256, NBATCH);
      scan_kernel<<<g, 256, 0, stream>>>(dtbuf, ubuf, dbl, alg, ybuf);
    }
    gate_kernel<<<(MROWS * D_INNER) / 256, 256, 0, stream>>>(ybuf, ubuf, xz, dsk);

    { // out_proj with residual into tokens: [1154,2048] x [1024,2048]^T += tokens
      dim3 g((D_MODEL + BN - 1) / BN, (MROWS + BM - 1) / BM);
      wmma_gemm_kernel<<<g, 256, 0, stream>>>(ybuf, D_INNER, opw, D_INNER,
                                              tokens, D_MODEL, nullptr,
                                              MROWS, D_MODEL, D_INNER, 3);
    }
  }
}
